// CustomBidirectionalGRU_76063870812447
// MI455X (gfx1250) — matrix operand order repaired and checked by reading, not compiled, not run on hardware
//
#include <hip/hip_runtime.h>

// ---------------------------------------------------------------------------
// Bidirectional 2-layer GRU for MI455X (gfx1250, wave32, WMMA)
//   B=32, T=1024, H=512, G=3H=1536
//   * gi = X @ w_ih^T + b_ih -> full-chip WMMA bf16 GEMM (time-parallel)
//   * recurrence -> persistent grid-synced scan; per wave: 3 gate n-tiles at
//     the SAME j-columns -> gate math from own accumulators, h in registers,
//     double-buffered bf16 h, ONE grid barrier/step, w_hh LDS-resident
//     (preloaded via global_load_async_to_lds_b128 + s_wait_asynccnt).
// ---------------------------------------------------------------------------

typedef __attribute__((ext_vector_type(16))) __bf16 v16bf;
typedef __attribute__((ext_vector_type(8)))  __bf16 bf16x8;
typedef __attribute__((ext_vector_type(8)))  float  v8f;
typedef __attribute__((ext_vector_type(4)))  int    v4i_;

#define B_   32
#define T_   1024
#define H_   512
#define G_   1536          /* 3H */
#define LDSK 520           /* padded K stride in bf16 (512 + 8) */
#define LDS_BYTES (192 * LDSK * 2) /* 192 cols x 520 bf16 = 199,680 B */

#if defined(__HIP_DEVICE_COMPILE__)
# if __has_builtin(__builtin_amdgcn_global_load_async_to_lds_b128) && \
     __has_builtin(__builtin_amdgcn_s_wait_asynccnt)
#  define USE_ASYNC_LDS 1
# endif
#endif
#ifndef USE_ASYNC_LDS
# define USE_ASYNC_LDS 0
#endif

// ---------------- fragment loaders (layouts per CDNA5 ISA) -------

__device__ inline v16bf cat8(bf16x8 lo, bf16x8 hi) {
  v16bf r;
#pragma unroll
  for (int i = 0; i < 8; ++i) { r[i] = lo[i]; r[8 + i] = hi[i]; }
  return r;
}

// A fragment 16x32 (MxK), bf16 row-major source, row stride `ld` elems.
__device__ inline v16bf load_a(const __bf16* base, int ld, int lane) {
  const int half = (lane >> 4) & 1, mr = lane & 15;
  const __bf16* p = base + (size_t)mr * ld + half * 8;
  bf16x8 lo = *reinterpret_cast<const bf16x8*>(p);
  bf16x8 hi = *reinterpret_cast<const bf16x8*>(p + 16);
  return cat8(lo, hi);
}

// A fragment with on-the-fly f32 -> bf16 conversion (for X).
__device__ inline v16bf load_a(const float* base, int ld, int lane) {
  const int half = (lane >> 4) & 1, mr = lane & 15;
  const float* p = base + (size_t)mr * ld + half * 8;
  v16bf r;
#pragma unroll
  for (int i = 0; i < 8; ++i) r[i] = (__bf16)p[i];
#pragma unroll
  for (int i = 0; i < 8; ++i) r[8 + i] = (__bf16)p[16 + i];
  return r;
}

// B fragment 32x16 (KxN) from row-major (N,K) weights in global memory.
__device__ inline v16bf load_b(const __bf16* w, int ldk, int nbase, int kbase,
                               int lane) {
  const __bf16* p = w + (size_t)(nbase + (lane & 15)) * ldk + kbase +
                    ((lane >> 4) & 1) * 8;
  bf16x8 lo = *reinterpret_cast<const bf16x8*>(p);
  bf16x8 hi = *reinterpret_cast<const bf16x8*>(p + 16);
  return cat8(lo, hi);
}

// B fragment from LDS-cached weight slice (col-major rows of LDSK bf16).
__device__ inline v16bf load_b_lds(const __bf16* lds, int cbase, int kbase,
                                   int lane) {
  const __bf16* p = lds + (size_t)(cbase + (lane & 15)) * LDSK + kbase +
                    ((lane >> 4) & 1) * 8;
  bf16x8 lo = *reinterpret_cast<const bf16x8*>(p);
  bf16x8 hi = *reinterpret_cast<const bf16x8*>(p + 16);
  return cat8(lo, hi);
}

__device__ inline float sigmoidf_(float x) {
  return 1.f / (1.f + __expf(-x));
}

// ---------------- small utility kernels -------------------------------------

__global__ void cvt_f32_bf16(const float* __restrict__ s, __bf16* __restrict__ d,
                             int n) {
  int i = blockIdx.x * blockDim.x + threadIdx.x;
  if (i < n) d[i] = (__bf16)s[i];
}

__global__ void init_bar_kernel(int* bar) {
  if (threadIdx.x < 8) bar[threadIdx.x] = 0;
}

// ---------------- gi = A @ W^T + bias  (M=32768, N=1536) --------------------

template <typename TA>
__global__ __launch_bounds__(256) void gi_gemm(const TA* __restrict__ A, int K,
                                               const __bf16* __restrict__ W,
                                               const float* __restrict__ bias,
                                               float* __restrict__ gi) {
  const int lane = threadIdx.x & 31;
  const int wave = threadIdx.x >> 5;
  const int m0 = (blockIdx.y * 8 + wave) * 16;
  const int n0 = blockIdx.x * 64;

  v8f acc[4];
#pragma unroll
  for (int j = 0; j < 4; ++j) acc[j] = {};

  for (int k0 = 0; k0 < K; k0 += 32) {
    v16bf a = load_a(A + (size_t)m0 * K + k0, K, lane);
#pragma unroll
    for (int j = 0; j < 4; ++j) {
      v16bf b = load_b(W, K, n0 + j * 16, k0, lane);
      acc[j] = __builtin_amdgcn_wmma_f32_16x16x32_bf16(
          false, a, false, b, (short)0, acc[j], false, false);
    }
  }

  const int half = (lane >> 4) & 1, col_l = lane & 15;
#pragma unroll
  for (int j = 0; j < 4; ++j) {
    const int col = n0 + j * 16 + col_l;
    const float bv = bias[col];
#pragma unroll
    for (int v = 0; v < 8; ++v) {
      const int row = m0 + half * 8 + v;
      gi[(size_t)row * G_ + col] = acc[j][v] + bv;
    }
  }
}

// ---------------- grid barrier (per-direction, 8 blocks) --------------------

__device__ inline void grid_barrier(int* cnt, volatile int* gen, int nblk) {
  __syncthreads();
  if (threadIdx.x == 0) {
    __threadfence();
    int g = *gen;
    if (atomicAdd(cnt, 1) == nblk - 1) {
      *cnt = 0;  // safe: all others already arrived
      __threadfence();
      atomicAdd((int*)gen, 1);
    } else {
      while (*gen == g) __builtin_amdgcn_s_sleep(2);
      __threadfence();
    }
  }
  __syncthreads();
}

// ---------------- persistent GRU scan ---------------------------------------
// 8 blocks x 256 threads per direction. Block owns j-cols [blk*64, blk*64+64),
// LDS-caches w_hh cols {g*512 + j} for g=0..2 (192 x 520 bf16). Wave (mt, jl)
// computes gates r/z/n for rows mt*16..+16, cols j0..j0+16 and updates h from
// its own accumulators; h kept in VGPRs, bf16 copy double-buffered in global.

struct ScanArgs {
  const float*  gi[2];
  const __bf16* whh[2];
  const float*  bhh[2];
  __bf16*       hbuf[2][2];  // [dir][phase], (32,512) bf16
  float*        yout;        // layer-1: d_out (B,T,2H) f32, else nullptr
  __bf16*       youtbf;      // layer-0: y0 bf16 (B,T,2H), else nullptr
  float*        hidden_out[2];
  int*          bar;         // {cnt,gen} x 2 dirs
  int           dir0;
};

__global__ __launch_bounds__(256) void gru_scan(ScanArgs p) {
  extern __shared__ __attribute__((aligned(16))) __bf16 lds[];
  const int dir   = p.dir0 + blockIdx.x / 8;
  const int blk   = blockIdx.x % 8;
  const int tid   = threadIdx.x;
  const int lane  = tid & 31;
  const int wave  = tid >> 5;
  const int mt    = wave & 1;          // M tile (batch rows 0-15 / 16-31)
  const int jl    = (wave >> 1) * 16;  // local j tile inside block
  const int jb    = blk * 64;          // block j base
  const int col_l = lane & 15;
  const int half  = (lane >> 4) & 1;

  const float*  gi  = p.gi[dir];
  const __bf16* whh = p.whh[dir];
  const float*  bhh = p.bhh[dir];
  __bf16* hb0 = p.hbuf[dir][0];
  __bf16* hb1 = p.hbuf[dir][1];
  int* cnt = p.bar + dir * 2;
  int* gen = cnt + 1;

  // ---- preload w_hh slice into LDS: lds[(g*64 + l)*LDSK + k] = whh[g*512+jb+l][k]
  for (int idx = tid; idx < 192 * 64; idx += 256) {
    const int c  = idx >> 6;          // 0..191 local column
    const int k8 = (idx & 63) << 3;   // k base (8 bf16 chunks)
    const int n  = (c >> 6) * H_ + jb + (c & 63);
    const __bf16* src = whh + (size_t)n * H_ + k8;
    __bf16*       dst = lds + (size_t)c * LDSK + k8;
#if USE_ASYNC_LDS
    __builtin_amdgcn_global_load_async_to_lds_b128(
        (__attribute__((address_space(1))) v4i_*)(void*)src,
        (__attribute__((address_space(3))) v4i_*)(void*)dst, 0, 0);
#else
    *reinterpret_cast<bf16x8*>(dst) = *reinterpret_cast<const bf16x8*>(src);
#endif
  }
#if USE_ASYNC_LDS
  __builtin_amdgcn_s_wait_asynccnt(0);
#endif
  __syncthreads();

  // ---- h = 0 (registers + phase-0 bf16 buffer)
  float hreg[8];
#pragma unroll
  for (int v = 0; v < 8; ++v) hreg[v] = 0.f;
  for (int i = blk * 256 + tid; i < B_ * H_; i += 2048) hb0[i] = (__bf16)0.f;
  __threadfence();
  grid_barrier(cnt, gen, 8);

  const int row_base = mt * 16 + half * 8;
  const int jcol     = jb + jl + col_l;  // global j in [0, 512)
  const float br_ = bhh[jcol], bz_ = bhh[H_ + jcol], bn_ = bhh[2 * H_ + jcol];

  for (int step = 0; step < T_; ++step) {
    const int t = dir ? (T_ - 1 - step) : step;
    const __bf16* hcur = (step & 1) ? hb1 : hb0;
    __bf16*       hnxt = (step & 1) ? hb0 : hb1;

    // ---- gh(r,z,n) for own (rows, j-cols) via WMMA, B from LDS ----
    v8f accr = {}, accz = {}, accn = {};
    for (int k0 = 0; k0 < H_; k0 += 32) {
      v16bf a = load_a(hcur + (size_t)(mt * 16) * H_ + k0, H_, lane);
      v16bf b0 = load_b_lds(lds, 0 * 64 + jl, k0, lane);
      accr = __builtin_amdgcn_wmma_f32_16x16x32_bf16(false, a, false, b0,
                                                     (short)0, accr, false, false);
      v16bf b1 = load_b_lds(lds, 1 * 64 + jl, k0, lane);
      accz = __builtin_amdgcn_wmma_f32_16x16x32_bf16(false, a, false, b1,
                                                     (short)0, accz, false, false);
      v16bf b2 = load_b_lds(lds, 2 * 64 + jl, k0, lane);
      accn = __builtin_amdgcn_wmma_f32_16x16x32_bf16(false, a, false, b2,
                                                     (short)0, accn, false, false);
    }

    // ---- gates + h update, all from own accumulators ----
#pragma unroll
    for (int v = 0; v < 8; ++v) {
      const int b = row_base + v;  // batch row
      const float* gib = gi + (size_t)(b * T_ + t) * G_;
      const float r = sigmoidf_(gib[jcol] + accr[v] + br_);
      const float z = sigmoidf_(gib[H_ + jcol] + accz[v] + bz_);
      const float n = tanhf(gib[2 * H_ + jcol] + r * (accn[v] + bn_));
      const float hnew = (1.f - z) * n + z * hreg[v];
      hreg[v] = hnew;
      hnxt[(size_t)b * H_ + jcol] = (__bf16)hnew;
      const size_t yo = (size_t)(b * T_ + t) * (2 * H_) + dir * H_ + jcol;
      if (p.yout)   p.yout[yo] = hnew;
      if (p.youtbf) p.youtbf[yo] = (__bf16)hnew;
    }
    __threadfence();
    grid_barrier(cnt, gen, 8);  // publish h for next step's A-fragment loads
  }

  float* ho = p.hidden_out[dir];
#pragma unroll
  for (int v = 0; v < 8; ++v) ho[(size_t)(row_base + v) * H_ + jcol] = hreg[v];
}

// ---------------- host-side orchestration -----------------------------------

extern "C" void kernel_launch(void* const* d_in, const int* in_sizes, int n_in,
                              void* d_out, int out_size, void* d_ws,
                              size_t ws_size, hipStream_t stream) {
  (void)in_sizes; (void)n_in; (void)out_size;

  const float* X = (const float*)d_in[0];
  const float* wih[4]  = {(const float*)d_in[1], (const float*)d_in[5],
                          (const float*)d_in[9], (const float*)d_in[13]};
  const float* bih[4]  = {(const float*)d_in[2], (const float*)d_in[6],
                          (const float*)d_in[10], (const float*)d_in[14]};
  const float* whhf[4] = {(const float*)d_in[3], (const float*)d_in[7],
                          (const float*)d_in[11], (const float*)d_in[15]};
  const float* bhh[4]  = {(const float*)d_in[4], (const float*)d_in[8],
                          (const float*)d_in[12], (const float*)d_in[16]};
  const int wih_n[4] = {G_ * H_, G_ * H_, G_ * 2 * H_, G_ * 2 * H_};

  char* ws = (char*)d_ws;
  size_t off = 0;
  auto take = [&](size_t bytes) -> char* {
    char* p = ws + off;
    off = (off + bytes + 255) & ~(size_t)255;
    return p;
  };

  __bf16* wihb[4]; __bf16* whhb[4];
  for (int i = 0; i < 4; ++i) wihb[i] = (__bf16*)take((size_t)wih_n[i] * 2);
  for (int i = 0; i < 4; ++i) whhb[i] = (__bf16*)take((size_t)G_ * H_ * 2);
  __bf16* y0bf = (__bf16*)take((size_t)B_ * T_ * 2 * H_ * 2);
  __bf16* hbuf[2][2];
  for (int d = 0; d < 2; ++d)
    for (int ph = 0; ph < 2; ++ph)
      hbuf[d][ph] = (__bf16*)take((size_t)B_ * H_ * 2);
  int* bar = (int*)take(256);
  const size_t gi_bytes = (size_t)B_ * T_ * G_ * 4;  // 201 MB
  float* gi0 = (float*)take(gi_bytes);
  const bool conc = (ws_size >= off + gi_bytes);  // room for 2nd gi buffer?
  float* gi1 = conc ? (float*)take(gi_bytes) : gi0;

  // allow > 64 KB dynamic LDS for the scan kernel
  (void)hipFuncSetAttribute((const void*)gru_scan,
                            hipFuncAttributeMaxDynamicSharedMemorySize,
                            LDS_BYTES);

  // convert weights f32 -> bf16 (deterministic, every call)
  for (int i = 0; i < 4; ++i) {
    cvt_f32_bf16<<<(wih_n[i] + 255) / 256, 256, 0, stream>>>(wih[i], wihb[i],
                                                             wih_n[i]);
    cvt_f32_bf16<<<(G_ * H_ + 255) / 256, 256, 0, stream>>>(whhf[i], whhb[i],
                                                            G_ * H_);
  }

  float* hid = (float*)d_out + (size_t)B_ * T_ * 2 * H_;
  const dim3 ggrid(G_ / 64, (B_ * T_) / 128);

  for (int layer = 0; layer < 2; ++layer) {
    const int wf = layer * 2, wr = layer * 2 + 1;
    const int K = layer ? 2 * H_ : H_;

    ScanArgs a = {};
    a.whh[0] = whhb[wf]; a.whh[1] = whhb[wr];
    a.bhh[0] = bhh[wf];  a.bhh[1] = bhh[wr];
    for (int d = 0; d < 2; ++d)
      for (int ph = 0; ph < 2; ++ph) a.hbuf[d][ph] = hbuf[d][ph];
    a.yout   = layer ? (float*)d_out : nullptr;
    a.youtbf = layer ? nullptr : y0bf;
    a.hidden_out[0] = hid + (size_t)(layer * 2 + 0) * B_ * H_;
    a.hidden_out[1] = hid + (size_t)(layer * 2 + 1) * B_ * H_;
    a.bar = bar;
    a.gi[0] = gi0; a.gi[1] = gi1;

    if (conc) {
      if (layer == 0) {
        gi_gemm<float><<<ggrid, 256, 0, stream>>>(X, K, wihb[wf], bih[wf], gi0);
        gi_gemm<float><<<ggrid, 256, 0, stream>>>(X, K, wihb[wr], bih[wr], gi1);
      } else {
        gi_gemm<__bf16><<<ggrid, 256, 0, stream>>>(y0bf, K, wihb[wf], bih[wf], gi0);
        gi_gemm<__bf16><<<ggrid, 256, 0, stream>>>(y0bf, K, wihb[wr], bih[wr], gi1);
      }
      init_bar_kernel<<<1, 32, 0, stream>>>(bar);
      a.dir0 = 0;
      gru_scan<<<16, 256, LDS_BYTES, stream>>>(a);
    } else {
      for (int dir = 0; dir < 2; ++dir) {
        const int wi = layer * 2 + dir;
        if (layer == 0)
          gi_gemm<float><<<ggrid, 256, 0, stream>>>(X, K, wihb[wi], bih[wi], gi0);
        else
          gi_gemm<__bf16><<<ggrid, 256, 0, stream>>>(y0bf, K, wihb[wi], bih[wi], gi0);
        init_bar_kernel<<<1, 32, 0, stream>>>(bar);
        a.gi[0] = gi0; a.gi[1] = gi0;
        a.dir0 = dir;
        gru_scan<<<8, 256, LDS_BYTES, stream>>>(a);
      }
    }
  }
}
